// Meta3D_54047868453328
// MI455X (gfx1250) — matrix operand order repaired and checked by reading, not compiled, not run on hardware
//
#include <hip/hip_runtime.h>

#define DIM    448
#define HEADS  8
#define KD     32
#define DV     128
#define NTOK   49
#define HQKV   1536
#define HID    1792
#define BATCH  1024
#define SCALE  0.17677669529663687f   // 1/sqrt(32)

typedef _Float16 half_t;
typedef __attribute__((ext_vector_type(16))) _Float16 v16h;
typedef __attribute__((ext_vector_type(8)))  float    v8f;

static __device__ inline v8f wmma16(v16h a, v16h b, v8f c) {
  return __builtin_amdgcn_wmma_f32_16x16x32_f16(false, a, false, b, (short)0, c, false, false);
}

// A-matrix fragment (16x32 f16), src row-major [rows][ld]. Branchless: rows >= maxrow
// are clamped to maxrow-1 (their garbage only lands in output rows we never store).
static __device__ inline v16h load_a_frag(const half_t* __restrict__ src, int ld,
                                          int mbase, int kb, int maxrow, int lane) {
  int row = mbase + (lane & 15);
  row = row < maxrow ? row : maxrow - 1;
  const int hf = lane >> 4;
  const half_t* p = src + (size_t)row * ld + kb + hf * 16;
  v16h a;
#pragma unroll
  for (int e = 0; e < 16; ++e) a[e] = p[e];
  return a;
}

// B-matrix fragment (32x16 f16) where B[k][n] = W[n][k], W row-major [N][ld].
// Contiguous 32B per lane.
static __device__ inline v16h load_b_frag(const half_t* __restrict__ W, int ld,
                                          int ncol, int kb, int lane) {
  const int col = ncol + (lane & 15);
  const int hf  = lane >> 4;
  const half_t* p = W + (size_t)col * ld + kb + hf * 16;
  v16h b;
#pragma unroll
  for (int e = 0; e < 16; ++e) b[e] = p[e];
  return b;
}

// K^T fragment: columns are tokens; clamp (garbage cols >=49 are masked by -1e30 bias).
static __device__ inline v16h load_bT_clamp(const half_t* __restrict__ base, int ld,
                                            int ncol, int kb, int maxcol, int lane) {
  int col = ncol + (lane & 15);
  col = col < maxcol ? col : maxcol - 1;
  const int hf = lane >> 4;
  const half_t* p = base + (size_t)col * ld + kb + hf * 16;
  v16h b;
#pragma unroll
  for (int e = 0; e < 16; ++e) b[e] = p[e];
  return b;
}

// P@V fragment: B's K dim walks token rows; clamp (P[:,k>=49] == 0 exactly).
static __device__ inline v16h load_b_rows_clamp(const half_t* __restrict__ base, int ld,
                                                int ncol, int kb, int maxrow, int lane) {
  const int col = ncol + (lane & 15);
  const int hf  = lane >> 4;
  v16h b;
#pragma unroll
  for (int e = 0; e < 16; ++e) {
    int k = kb + hf * 16 + e;
    k = k < maxrow ? k : maxrow - 1;
    b[e] = base[(size_t)k * ld + col];
  }
  return b;
}

static __device__ inline float wave_sum(float v) {
#pragma unroll
  for (int o = 16; o > 0; o >>= 1) v += __shfl_xor(v, o, 32);
  return v;
}

// ---------------------------------------------------------------- prep: weights f32->f16 + bias gather
__global__ void prep_kernel(const float* __restrict__ qkv_w, const float* __restrict__ proj_w,
                            const float* __restrict__ fc1_w, const float* __restrict__ fc2_w,
                            const float* __restrict__ abias, const int* __restrict__ bidx,
                            half_t* __restrict__ wq, half_t* __restrict__ wp,
                            half_t* __restrict__ w1, half_t* __restrict__ w2,
                            float* __restrict__ biasP) {
  const int NQ = HQKV * DIM, NP = DIM * (HEADS * DV), N1 = HID * DIM, N2 = DIM * HID;
  const int total = NQ + NP + N1 + N2;
  for (int i = blockIdx.x * blockDim.x + threadIdx.x; i < total; i += gridDim.x * blockDim.x) {
    if (i < NQ)                wq[i] = (half_t)qkv_w[i];
    else if (i < NQ + NP)      wp[i - NQ] = (half_t)proj_w[i - NQ];
    else if (i < NQ + NP + N1) w1[i - NQ - NP] = (half_t)fc1_w[i - NQ - NP];
    else                       w2[i - NQ - NP - N1] = (half_t)fc2_w[i - NQ - NP - N1];
  }
  for (int i = blockIdx.x * blockDim.x + threadIdx.x; i < HEADS * 64 * 64; i += gridDim.x * blockDim.x) {
    const int h = i >> 12, r = (i >> 6) & 63, c = i & 63;
    biasP[i] = (r < NTOK && c < NTOK) ? abias[h * 49 + bidx[r * NTOK + c]] : -1.0e30f;
  }
}

// ---------------------------------------------------------------- layernorm: one wave per token
__global__ void ln_kernel(const float* __restrict__ in, const float* __restrict__ g,
                          const float* __restrict__ b, half_t* __restrict__ out, int rows) {
  const int wid  = (blockIdx.x * blockDim.x + threadIdx.x) >> 5;
  const int lane = threadIdx.x & 31;
  if (wid >= rows) return;
  const float* x = in + (size_t)wid * DIM;
  float vals[14];
  float s = 0.f;
#pragma unroll
  for (int i = 0; i < 14; ++i) { vals[i] = x[lane + 32 * i]; s += vals[i]; }
  s = wave_sum(s);
  const float mu = s * (1.0f / DIM);
  float vs = 0.f;
#pragma unroll
  for (int i = 0; i < 14; ++i) { const float d = vals[i] - mu; vs += d * d; }
  vs = wave_sum(vs);
  const float rstd = rsqrtf(vs * (1.0f / DIM) + 1e-5f);
  half_t* o = out + (size_t)wid * DIM;
#pragma unroll
  for (int i = 0; i < 14; ++i) {
    const int c = lane + 32 * i;
    o[c] = (half_t)((vals[i] - mu) * rstd * g[c] + b[c]);
  }
}

// ---------------------------------------------------------------- QKV GEMM: [49x448]x[448x1536]
__global__ __launch_bounds__(256) void qkv_gemm_kernel(const half_t* __restrict__ hbuf,
                                                       const half_t* __restrict__ wq,
                                                       const float* __restrict__ qkv_b,
                                                       half_t* __restrict__ qkv) {
  const int b = blockIdx.x, nbase0 = blockIdx.y * 384;   // 4 N-splits of 384
  const int lane = threadIdx.x & 31, wave = threadIdx.x >> 5;
  const half_t* A = hbuf + (size_t)b * NTOK * DIM;
  half_t* O = qkv + (size_t)b * NTOK * HQKV;
  auto epi = [&](v8f c, int mb, int nb) {
    const int col = nb + (lane & 15), hf = lane >> 4;
    const float bias = qkv_b[col];
#pragma unroll
    for (int v = 0; v < 8; ++v) {
      const int r = mb + v + 8 * hf;
      if (r < NTOK) O[(size_t)r * HQKV + col] = (half_t)(c[v] + bias);
    }
  };
  for (int t = wave; t < 24; t += 8) {               // 2 m-macros x 12 n-macros (32x32 each)
    const int mb = (t & 1) * 32, nb = nbase0 + (t >> 1) * 32;
    v8f c00 = {}, c01 = {}, c10 = {}, c11 = {};
    for (int ks = 0; ks < 14; ++ks) {
      __builtin_prefetch(wq + (size_t)(nb + (lane & 15)) * DIM + ks * 32 + 64, 0, 1);
      v16h a0 = load_a_frag(A, DIM, mb,      ks * 32, NTOK, lane);
      v16h a1 = load_a_frag(A, DIM, mb + 16, ks * 32, NTOK, lane);
      v16h b0 = load_b_frag(wq, DIM, nb,      ks * 32, lane);
      v16h b1 = load_b_frag(wq, DIM, nb + 16, ks * 32, lane);
      c00 = wmma16(a0, b0, c00); c01 = wmma16(a0, b1, c01);
      c10 = wmma16(a1, b0, c10); c11 = wmma16(a1, b1, c11);
    }
    epi(c00, mb, nb); epi(c01, mb, nb + 16);
    epi(c10, mb + 16, nb); epi(c11, mb + 16, nb + 16);
  }
}

// ---------------------------------------------------------------- attention per (batch, head)
__global__ __launch_bounds__(128) void attn_kernel(const half_t* __restrict__ qkv,
                                                   const float* __restrict__ biasP,
                                                   half_t* __restrict__ aout) {
  __shared__ float  S[64 * 64];
  __shared__ half_t P[64 * 64];
  const int b = blockIdx.x / HEADS, h = blockIdx.x % HEADS;
  const int lane = threadIdx.x & 31, wave = threadIdx.x >> 5;  // 4 waves
  const half_t* base = qkv + (size_t)b * NTOK * HQKV + h * (KD + KD + DV);
  const half_t* qp = base;            // q = +0..31   (row stride HQKV)
  const half_t* kp = base + KD;       // k = +32..63
  const half_t* vp = base + 2 * KD;   // v = +64..191
  const float* bp = biasP + h * 64 * 64;

  // S = (q @ k^T) * scale + bias : one 32x32 macro-tile per wave, K=32
  {
    const int mb = (wave & 1) * 32, nb = (wave >> 1) * 32;
    v16h a0 = load_a_frag(qp, HQKV, mb,      0, NTOK, lane);
    v16h a1 = load_a_frag(qp, HQKV, mb + 16, 0, NTOK, lane);
    v16h b0 = load_bT_clamp(kp, HQKV, nb,      0, NTOK, lane);
    v16h b1 = load_bT_clamp(kp, HQKV, nb + 16, 0, NTOK, lane);
    v8f c00 = {}, c01 = {}, c10 = {}, c11 = {};
    c00 = wmma16(a0, b0, c00); c01 = wmma16(a0, b1, c01);
    c10 = wmma16(a1, b0, c10); c11 = wmma16(a1, b1, c11);
    auto sepi = [&](v8f c, int mbs, int nbs) {
      const int col = nbs + (lane & 15), hf = lane >> 4;
#pragma unroll
      for (int v = 0; v < 8; ++v) {
        const int r = mbs + v + 8 * hf;
        S[r * 64 + col] = c[v] * SCALE + bp[r * 64 + col];
      }
    };
    sepi(c00, mb, nb); sepi(c01, mb, nb + 16);
    sepi(c10, mb + 16, nb); sepi(c11, mb + 16, nb + 16);
  }
  __syncthreads();

  // row softmax (padded cols carry -1e30 bias -> exp()==0 exactly)
  if (threadIdx.x < 64) {
    const int r = threadIdx.x;
    float mx = -3.0e38f;
    for (int c = 0; c < 64; ++c) mx = fmaxf(mx, S[r * 64 + c]);
    float sum = 0.f;
    for (int c = 0; c < 64; ++c) sum += __expf(S[r * 64 + c] - mx);
    const float inv = 1.0f / sum;
    for (int c = 0; c < 64; ++c) P[r * 64 + c] = (half_t)(__expf(S[r * 64 + c] - mx) * inv);
  }
  __syncthreads();

  // out = P(64x64) @ v(64x128): 2x4 macro-tiles of 32x32, K=64
  half_t* O = aout + (size_t)b * NTOK * (HEADS * DV) + h * DV;
  for (int t = wave; t < 8; t += 4) {
    const int mb = (t & 1) * 32, nb = (t >> 1) * 32;
    v8f c00 = {}, c01 = {}, c10 = {}, c11 = {};
#pragma unroll
    for (int ks = 0; ks < 2; ++ks) {
      v16h a0 = load_a_frag(P, 64, mb,      ks * 32, 64, lane);
      v16h a1 = load_a_frag(P, 64, mb + 16, ks * 32, 64, lane);
      v16h b0 = load_b_rows_clamp(vp, HQKV, nb,      ks * 32, NTOK, lane);
      v16h b1 = load_b_rows_clamp(vp, HQKV, nb + 16, ks * 32, NTOK, lane);
      c00 = wmma16(a0, b0, c00); c01 = wmma16(a0, b1, c01);
      c10 = wmma16(a1, b0, c10); c11 = wmma16(a1, b1, c11);
    }
    auto oepi = [&](v8f c, int mbs, int nbs) {
      const int col = nbs + (lane & 15), hf = lane >> 4;
#pragma unroll
      for (int v = 0; v < 8; ++v) {
        const int r = mbs + v + 8 * hf;
        if (r < NTOK) O[(size_t)r * (HEADS * DV) + col] = (half_t)c[v];
      }
    };
    oepi(c00, mb, nb); oepi(c01, mb, nb + 16);
    oepi(c10, mb + 16, nb); oepi(c11, mb + 16, nb + 16);
  }
}

// ---------------------------------------------------------------- proj + layerscale residual
__global__ __launch_bounds__(256) void proj_res_kernel(const half_t* __restrict__ aout,
                                                       const half_t* __restrict__ wp,
                                                       const float* __restrict__ proj_b,
                                                       const float* __restrict__ ls1,
                                                       const float* __restrict__ x,
                                                       float* __restrict__ ybuf) {
  const int b = blockIdx.x;
  const int lane = threadIdx.x & 31, wave = threadIdx.x >> 5;
  const half_t* A = aout + (size_t)b * NTOK * 1024;
  auto epi = [&](v8f c, int mb, int nb) {
    const int col = nb + (lane & 15), hf = lane >> 4;
    const float bias = proj_b[col], gls = ls1[col];
#pragma unroll
    for (int v = 0; v < 8; ++v) {
      const int r = mb + v + 8 * hf;
      if (r < NTOK) {
        const size_t idx = ((size_t)b * NTOK + r) * DIM + col;
        ybuf[idx] = x[idx] + (c[v] + bias) * gls;
      }
    }
  };
  for (int t = wave; t < 28; t += 8) {               // 2 x 14 macro-tiles (32x32), K=1024
    const int mb = (t & 1) * 32, nb = (t >> 1) * 32;
    v8f c00 = {}, c01 = {}, c10 = {}, c11 = {};
    for (int ks = 0; ks < 32; ++ks) {
      __builtin_prefetch(wp + (size_t)(nb + (lane & 15)) * 1024 + ks * 32 + 64, 0, 1);
      v16h a0 = load_a_frag(A, 1024, mb,      ks * 32, NTOK, lane);
      v16h a1 = load_a_frag(A, 1024, mb + 16, ks * 32, NTOK, lane);
      v16h b0 = load_b_frag(wp, 1024, nb,      ks * 32, lane);
      v16h b1 = load_b_frag(wp, 1024, nb + 16, ks * 32, lane);
      c00 = wmma16(a0, b0, c00); c01 = wmma16(a0, b1, c01);
      c10 = wmma16(a1, b0, c10); c11 = wmma16(a1, b1, c11);
    }
    epi(c00, mb, nb); epi(c01, mb, nb + 16);
    epi(c10, mb + 16, nb); epi(c11, mb + 16, nb + 16);
  }
}

// ---------------------------------------------------------------- fc1 + exact GELU
__global__ __launch_bounds__(256) void fc1_gelu_kernel(const half_t* __restrict__ h2,
                                                       const half_t* __restrict__ w1,
                                                       const float* __restrict__ fc1_b,
                                                       half_t* __restrict__ gbuf) {
  const int b = blockIdx.x, nbase0 = blockIdx.y * 448;   // 4 N-splits of 448
  const int lane = threadIdx.x & 31, wave = threadIdx.x >> 5;
  const half_t* A = h2 + (size_t)b * NTOK * DIM;
  half_t* O = gbuf + (size_t)b * NTOK * HID;
  auto epi = [&](v8f c, int mb, int nb) {
    const int col = nb + (lane & 15), hf = lane >> 4;
    const float bias = fc1_b[col];
#pragma unroll
    for (int v = 0; v < 8; ++v) {
      const int r = mb + v + 8 * hf;
      if (r < NTOK) {
        const float tv = c[v] + bias;
        O[(size_t)r * HID + col] = (half_t)(0.5f * tv * (1.0f + erff(tv * 0.70710678118654752f)));
      }
    }
  };
  for (int t = wave; t < 28; t += 8) {               // 2 x 14 macro-tiles, K=448
    const int mb = (t & 1) * 32, nb = nbase0 + (t >> 1) * 32;
    v8f c00 = {}, c01 = {}, c10 = {}, c11 = {};
    for (int ks = 0; ks < 14; ++ks) {
      __builtin_prefetch(w1 + (size_t)(nb + (lane & 15)) * DIM + ks * 32 + 64, 0, 1);
      v16h a0 = load_a_frag(A, DIM, mb,      ks * 32, NTOK, lane);
      v16h a1 = load_a_frag(A, DIM, mb + 16, ks * 32, NTOK, lane);
      v16h b0 = load_b_frag(w1, DIM, nb,      ks * 32, lane);
      v16h b1 = load_b_frag(w1, DIM, nb + 16, ks * 32, lane);
      c00 = wmma16(a0, b0, c00); c01 = wmma16(a0, b1, c01);
      c10 = wmma16(a1, b0, c10); c11 = wmma16(a1, b1, c11);
    }
    epi(c00, mb, nb); epi(c01, mb, nb + 16);
    epi(c10, mb + 16, nb); epi(c11, mb + 16, nb + 16);
  }
}

// ---------------------------------------------------------------- fc2 + layerscale residual -> out
__global__ __launch_bounds__(256) void fc2_res_kernel(const half_t* __restrict__ gbuf,
                                                      const half_t* __restrict__ w2,
                                                      const float* __restrict__ fc2_b,
                                                      const float* __restrict__ ls2,
                                                      const float* __restrict__ ybuf,
                                                      float* __restrict__ out) {
  const int b = blockIdx.x;
  const int lane = threadIdx.x & 31, wave = threadIdx.x >> 5;
  const half_t* A = gbuf + (size_t)b * NTOK * HID;
  auto epi = [&](v8f c, int mb, int nb) {
    const int col = nb + (lane & 15), hf = lane >> 4;
    const float bias = fc2_b[col], gls = ls2[col];
#pragma unroll
    for (int v = 0; v < 8; ++v) {
      const int r = mb + v + 8 * hf;
      if (r < NTOK) {
        const size_t idx = ((size_t)b * NTOK + r) * DIM + col;
        out[idx] = ybuf[idx] + (c[v] + bias) * gls;
      }
    }
  };
  for (int t = wave; t < 28; t += 8) {               // 2 x 14 macro-tiles, K=1792
    const int mb = (t & 1) * 32, nb = (t >> 1) * 32;
    v8f c00 = {}, c01 = {}, c10 = {}, c11 = {};
    for (int ks = 0; ks < 56; ++ks) {
      __builtin_prefetch(w2 + (size_t)(nb + (lane & 15)) * HID + ks * 32 + 64, 0, 1);
      v16h a0 = load_a_frag(A, HID, mb,      ks * 32, NTOK, lane);
      v16h a1 = load_a_frag(A, HID, mb + 16, ks * 32, NTOK, lane);
      v16h b0 = load_b_frag(w2, HID, nb,      ks * 32, lane);
      v16h b1 = load_b_frag(w2, HID, nb + 16, ks * 32, lane);
      c00 = wmma16(a0, b0, c00); c01 = wmma16(a0, b1, c01);
      c10 = wmma16(a1, b0, c10); c11 = wmma16(a1, b1, c11);
    }
    epi(c00, mb, nb); epi(c01, mb, nb + 16);
    epi(c10, mb + 16, nb); epi(c11, mb + 16, nb + 16);
  }
}

// ================================================================ host
extern "C" void kernel_launch(void* const* d_in, const int* in_sizes, int n_in,
                              void* d_out, int out_size, void* d_ws, size_t ws_size,
                              hipStream_t stream) {
  const float* x      = (const float*)d_in[0];
  const float* qkv_w  = (const float*)d_in[1];
  const float* qkv_b  = (const float*)d_in[2];
  const float* proj_w = (const float*)d_in[3];
  const float* proj_b = (const float*)d_in[4];
  const float* fc1_w  = (const float*)d_in[5];
  const float* fc1_b  = (const float*)d_in[6];
  const float* fc2_w  = (const float*)d_in[7];
  const float* fc2_b  = (const float*)d_in[8];
  const float* n1g    = (const float*)d_in[9];
  const float* n1b    = (const float*)d_in[10];
  const float* n2g    = (const float*)d_in[11];
  const float* n2b    = (const float*)d_in[12];
  const float* ls1    = (const float*)d_in[13];
  const float* ls2    = (const float*)d_in[14];
  const float* abias  = (const float*)d_in[15];
  const int*   bidx   = (const int*)d_in[16];
  float* out = (float*)d_out;

  char* ws = (char*)d_ws;
  size_t off = 0;
  auto carve = [&](size_t bytes) -> char* {
    char* p = ws + off;
    off += (bytes + 255) & ~(size_t)255;
    return p;
  };
  half_t* wq    = (half_t*)carve((size_t)HQKV * DIM * 2);
  half_t* wp    = (half_t*)carve((size_t)DIM * 1024 * 2);
  half_t* w1    = (half_t*)carve((size_t)HID * DIM * 2);
  half_t* w2    = (half_t*)carve((size_t)DIM * HID * 2);
  float*  biasP = (float*)carve((size_t)HEADS * 64 * 64 * 4);
  half_t* hbuf  = (half_t*)carve((size_t)BATCH * NTOK * DIM * 2);   // LN1 out, reused for LN2 out
  float*  ybuf  = (float*)carve((size_t)BATCH * NTOK * DIM * 4);    // post-attn residual (f32)
  half_t* aout  = (half_t*)carve((size_t)BATCH * NTOK * 1024 * 2);  // attention out
  half_t* big   = (half_t*)carve((size_t)BATCH * NTOK * HID * 2);   // qkv (B*49*1536) then gelu (B*49*1792)
  half_t* qkvb  = big;
  half_t* gbuf  = big;

  const int rows = BATCH * NTOK;

  prep_kernel<<<4096, 256, 0, stream>>>(qkv_w, proj_w, fc1_w, fc2_w, abias, bidx,
                                        wq, wp, w1, w2, biasP);
  ln_kernel<<<(rows * 32 + 255) / 256, 256, 0, stream>>>(x, n1g, n1b, hbuf, rows);
  qkv_gemm_kernel<<<dim3(BATCH, 4), 256, 0, stream>>>(hbuf, wq, qkv_b, qkvb);
  attn_kernel<<<BATCH * HEADS, 128, 0, stream>>>(qkvb, biasP, aout);
  proj_res_kernel<<<BATCH, 256, 0, stream>>>(aout, wp, proj_b, ls1, x, ybuf);
  ln_kernel<<<(rows * 32 + 255) / 256, 256, 0, stream>>>(ybuf, n2g, n2b, hbuf, rows);
  fc1_gelu_kernel<<<dim3(BATCH, 4), 256, 0, stream>>>(hbuf, w1, fc1_b, gbuf);
  fc2_res_kernel<<<BATCH, 256, 0, stream>>>(gbuf, w2, fc2_b, ls2, ybuf, out);
}
